// LSTNet_Att_10685878632622
// MI455X (gfx1250) — matrix operand order repaired and checked by reading, not compiled, not run on hardware
//
#include <hip/hip_runtime.h>
#include <hip/hip_bf16.h>
#include <math.h>

// LSTNet+attention for MI455X (gfx1250, wave32, WMMA + async-LDS).
// bf16 WMMA (fp32 accum) for conv im2col GEMM and all LSTM step GEMMs;
// GLOBAL_LOAD_ASYNC_TO_LDS_B128 tile staging with LDS double-buffering in the
// latency-bound sequential LSTM steps; fp32 VALU for gates/softmax/head.
// Workspace requirement: ~330 MB.

#define DI __device__ __forceinline__

typedef __attribute__((ext_vector_type(16))) __bf16        v16bf;
typedef __attribute__((ext_vector_type(8)))  float         v8f;
typedef __attribute__((ext_vector_type(4)))  unsigned int  u32x4;
typedef __attribute__((ext_vector_type(4)))  int           v4i;

#define AS1 __attribute__((address_space(1)))
#define AS3 __attribute__((address_space(3)))

#if defined(__has_builtin)
#if __has_builtin(__builtin_amdgcn_global_load_async_to_lds_b128)
#define HAVE_ASYNC_COPY 1
#endif
#endif

static constexpr int Bb = 512;   // batch
static constexpr int Ww = 168;   // time
static constexpr int Dd = 321;   // feature
static constexpr int Kk = 6;     // conv taps
static constexpr int HC = 512;
static constexpr int HG = 512;
static constexpr int AR = 24;
static constexpr int Ts = Ww - 1;        // 167 LSTM steps
static constexpr int KT = Kk * Dd;       // 1926 im2col depth
static constexpr int KTP = 1952;         // padded conv-weight row stride (16B-aligned)
static constexpr int WD = Ww * Dd;       // x row-slice stride per batch

// float -> bf16 bits, round-to-nearest-even
DI unsigned short f2bf(float f) {
    unsigned u = __float_as_uint(f);
    unsigned r = 0x7FFFu + ((u >> 16) & 1u);
    return (unsigned short)((u + r) >> 16);
}

DI v8f wmma_bf16(v16bf a, v16bf b, v8f c) {
    return __builtin_amdgcn_wmma_f32_16x16x32_bf16(
        false, a, false, b, (short)0, c, false, false);
}

// 16-byte global -> LDS copy. Async (ASYNCcnt-tracked) when the toolchain
// exposes the gfx1250 builtin, else a vectorized sync copy.
DI void copy16B_g2l(unsigned short* dst_lds, const unsigned short* src_g) {
#ifdef HAVE_ASYNC_COPY
    __builtin_amdgcn_global_load_async_to_lds_b128(
        (AS1 v4i*)(AS1 void*)const_cast<unsigned short*>(src_g),
        (AS3 v4i*)(AS3 void*)dst_lds, 0, 0);
#else
    *(u32x4*)dst_lds = *(const u32x4*)src_g;
#endif
}

DI void async_wait0() {
#ifdef HAVE_ASYNC_COPY
#if __has_builtin(__builtin_amdgcn_s_wait_asynccnt)
    __builtin_amdgcn_s_wait_asynccnt(0);
#else
    asm volatile("s_wait_asynccnt 0" ::: "memory");
#endif
#endif
}

// A fragment (16x32 bf16) from LDS tile [row][k], k-stride 32.
DI v16bf load_frag_a(const unsigned short* At, int mBase, int lane) {
    const int half = lane >> 4, mr = lane & 15;
    const unsigned short* p = At + (mBase + mr) * 32 + (half ? 8 : 0);
    union { v16bf v; u32x4 q[2]; } f;
    f.q[0] = *(const u32x4*)(p);
    f.q[1] = *(const u32x4*)(p + 16);
    return f.v;
}

// B fragment (32x16 bf16) from LDS tile [n][k], k-stride 32.
DI v16bf load_frag_b(const unsigned short* Bt, int nBase, int lane) {
    const int half = lane >> 4, nr = lane & 15;
    const unsigned short* p = Bt + (nBase + nr) * 32 + (half ? 8 : 0);
    union { v16bf v; u32x4 q[2]; } f;
    f.q[0] = *(const u32x4*)(p);
    f.q[1] = *(const u32x4*)(p + 16);
    return f.v;
}

// ---------------------------------------------------------------- utilities
__global__ __launch_bounds__(256) void cvt_f32_bf16(const float* __restrict__ in,
                                                    unsigned short* __restrict__ out, int n) {
    int i = blockIdx.x * 256 + threadIdx.x;
    if (i < n) out[i] = f2bf(in[i]);
}

// conv_w (HC, KT) f32 -> (HC, KTP) bf16 with zero-padded row tails
__global__ __launch_bounds__(256) void cvt_convw(const float* __restrict__ in,
                                                 unsigned short* __restrict__ out) {
    int i = blockIdx.x * 256 + threadIdx.x;
    if (i >= HC * KTP) return;
    int hc = i / KTP, kk = i - hc * KTP;
    out[i] = (kk < KT) ? f2bf(in[(size_t)hc * KT + kk]) : (unsigned short)0;
}

__global__ __launch_bounds__(256) void zero_f32(float* __restrict__ p, int n) {
    int i = blockIdx.x * 256 + threadIdx.x;
    if (i < n) p[i] = 0.0f;
}

__global__ __launch_bounds__(256) void zero_bf16(unsigned short* __restrict__ p, int n) {
    int i = blockIdx.x * 256 + threadIdx.x;
    if (i < n) p[i] = 0;
}

// ------------------------------------------------- conv im2col GEMM (WMMA)
// Because the conv window is time-contiguous over full D, im2col row m=(t,b)
// is the contiguous slice x_flat[b*W*D + (t-5)*D + kk], kk in [0,1926).
// Tile: 128 rows x 64 cols; 8 waves (4x2), wave = 32x32 (2x2 WMMA tiles).
__global__ __launch_bounds__(256) void conv_gemm_kernel(
    const unsigned short* __restrict__ xbf,   // (B,W,D) bf16
    const unsigned short* __restrict__ cwbf,  // (HC, KTP) bf16, zero-padded rows
    const float* __restrict__ conv_b,         // (HC)
    unsigned short* __restrict__ gbf)         // (Ts,B,HC) bf16
{
    __shared__ __attribute__((aligned(16))) unsigned short At[128 * 32];
    __shared__ __attribute__((aligned(16))) unsigned short Bt[64 * 32];

    const int rowBase = blockIdx.x * 128;          // 128 | 512 -> single t per tile
    const int colBase = blockIdx.y * 64;
    const int t  = rowBase / Bb;
    const int b0 = rowBase % Bb;
    const int baseT = (t - (Kk - 1)) * Dd;         // signed start offset in x row-slice
    const int tid  = threadIdx.x;
    const int lane = tid & 31;
    const int wave = tid >> 5;
    const int wrow = wave >> 1;                    // 0..3
    const int wcol = wave & 1;                     // 0..1

    v8f acc[2][2] = {};

    for (int k0 = 0; k0 < KT; k0 += 32) {
        __syncthreads();
        // ---- A tile: contiguous slice of x per row (no division, no gather)
        if (baseT >= 0 && k0 + 32 <= KT) {         // uniform fast path (t>=5)
            for (int i = tid; i < 128 * 32; i += 256) {
                int r = i >> 5, kk = (i & 31) + k0;
                At[i] = xbf[(size_t)(b0 + r) * WD + baseT + kk];
            }
        } else {                                    // left zero-pad / K tail
            for (int i = tid; i < 128 * 32; i += 256) {
                int r = i >> 5, kk = (i & 31) + k0;
                int gofs = baseT + kk;
                unsigned short v = 0;
                if (kk < KT && gofs >= 0)
                    v = xbf[(size_t)(b0 + r) * WD + gofs];
                At[i] = v;
            }
        }
        // ---- B tile: padded weight rows -> unchecked async b128 copies
        {
            int n = tid >> 2, gk = (tid & 3) * 8;   // 256 groups of 8 shorts
            copy16B_g2l(Bt + n * 32 + gk,
                        cwbf + (size_t)(colBase + n) * KTP + k0 + gk);
        }
        async_wait0();
        __syncthreads();

        v16bf a0  = load_frag_a(At, wrow * 32,      lane);
        v16bf a1  = load_frag_a(At, wrow * 32 + 16, lane);
        v16bf bq0 = load_frag_b(Bt, wcol * 32,      lane);
        v16bf bq1 = load_frag_b(Bt, wcol * 32 + 16, lane);
        acc[0][0] = wmma_bf16(a0, bq0, acc[0][0]);
        acc[0][1] = wmma_bf16(a0, bq1, acc[0][1]);
        acc[1][0] = wmma_bf16(a1, bq0, acc[1][0]);
        acc[1][1] = wmma_bf16(a1, bq1, acc[1][1]);
    }

    const int half = lane >> 4, nr = lane & 15;
    for (int mi = 0; mi < 2; ++mi)
        for (int ni = 0; ni < 2; ++ni)
            for (int r = 0; r < 8; ++r) {
                int row = wrow * 32 + mi * 16 + (half ? 8 : 0) + r;   // 0..127
                int col = colBase + wcol * 32 + ni * 16 + nr;         // hc
                float v = acc[mi][ni][r] + conv_b[col];
                v = v > 0.0f ? v : 0.0f;
                gbf[(size_t)(rowBase + row) * HC + col] = f2bf(v);
            }
}

// --------------------------------------------------- one LSTM step (WMMA)
// Tile: 64 batch x 64 hg x 4 gates per workgroup. 8 waves: gate = w>>1,
// row-half = w&1. Unified 32-k-step loop (w_ih phase then w_hh phase) with
// async double-buffered LDS tiles so global latency hides under the WMMAs.
// Buffer layout (in shorts): buf i at base + i*10240; A = 64*32 = 2048 shorts,
// then B = 4*64*32 = 8192 shorts.
DI void lstm_fill(int ks, unsigned short* At, unsigned short* Bt,
                  const unsigned short* gt, const unsigned short* hin,
                  const unsigned short* wih, const unsigned short* whh,
                  int bBase, int hgBase, int tid) {
    const unsigned short* Ag = (ks & 16) ? hin : gt;
    const unsigned short* Wg = (ks & 16) ? whh : wih;
    const int k0 = (ks & 15) * 32;
    // A: 64x32 = 256 groups of 8 shorts (one per thread)
    {
        int r = tid >> 2, gk = (tid & 3) * 8;
        copy16B_g2l(At + r * 32 + gk, Ag + (size_t)(bBase + r) * 512 + k0 + gk);
    }
    // B: 4 gates x 64 x 32 = 1024 groups (four per thread)
    for (int g = tid; g < 1024; g += 256) {
        int gate = g >> 8, n = (g >> 2) & 63, gk = (g & 3) * 8;
        copy16B_g2l(Bt + g * 8,
                    Wg + (size_t)(gate * 512 + hgBase + n) * 512 + k0 + gk);
    }
}

__global__ __launch_bounds__(256) void lstm_step_kernel(
    const unsigned short* __restrict__ gt,    // (B,HC) bf16 step input
    const unsigned short* __restrict__ hin,   // (B,HG) bf16
    const unsigned short* __restrict__ wih,   // (4HG,HC) bf16
    const unsigned short* __restrict__ whh,   // (4HG,HG) bf16
    const float* __restrict__ b_ih,
    const float* __restrict__ b_hh,
    float* __restrict__ cbuf,                 // (B,HG) f32 in/out
    float* __restrict__ hout_f,               // (B,HG) f32  (out[t] slice / h_final)
    unsigned short* __restrict__ hout_bf)     // (B,HG) bf16 (next step input)
{
    __shared__ __attribute__((aligned(16))) unsigned char smem[4 * 64 * 64 * 4]; // 64 KB
    unsigned short* tbase = (unsigned short*)smem;   // double-buffered tiles
    float* gsm = (float*)smem;                       // 4*64*64 f32 (aliased after GEMM)

    const int bBase  = blockIdx.x * 64;
    const int hgBase = blockIdx.y * 64;
    const int tid  = threadIdx.x;
    const int lane = tid & 31;
    const int wave = tid >> 5;
    const int gsel  = wave >> 1;     // 0..3 gate
    const int rhalf = wave & 1;      // 0..1 row half

    v8f acc[2][4] = {};

    lstm_fill(0, tbase, tbase + 2048, gt, hin, wih, whh, bBase, hgBase, tid);
    async_wait0();
    __syncthreads();

    for (int ks = 0; ks < 32; ++ks) {
        const int cur = ks & 1;
        unsigned short* Atc = tbase + cur * 10240;
        if (ks + 1 < 32) {   // prefetch next k-step into the other buffer
            unsigned short* Atn = tbase + (cur ^ 1) * 10240;
            lstm_fill(ks + 1, Atn, Atn + 2048,
                      gt, hin, wih, whh, bBase, hgBase, tid);
        }

        v16bf a0 = load_frag_a(Atc, rhalf * 32,      lane);
        v16bf a1 = load_frag_a(Atc, rhalf * 32 + 16, lane);
        const unsigned short* Bg = Atc + 2048 + gsel * 64 * 32;
        for (int nb = 0; nb < 4; ++nb) {
            v16bf bb = load_frag_b(Bg, nb * 16, lane);
            acc[0][nb] = wmma_bf16(a0, bb, acc[0][nb]);
            acc[1][nb] = wmma_bf16(a1, bb, acc[1][nb]);
        }

        async_wait0();     // prefetched tiles have landed
        __syncthreads();   // everyone done reading `cur` before next overwrite
    }

    // scatter gate tiles to LDS: gsm[(gate*64 + row)*64 + col]
    {
        const int half = lane >> 4, nr = lane & 15;
        for (int mi = 0; mi < 2; ++mi)
            for (int nb = 0; nb < 4; ++nb)
                for (int r = 0; r < 8; ++r) {
                    int row = rhalf * 32 + mi * 16 + (half ? 8 : 0) + r;
                    int col = nb * 16 + nr;
                    gsm[(gsel * 64 + row) * 64 + col] = acc[mi][nb][r];
                }
    }
    __syncthreads();

    // fused gate nonlinearity + cell update (fp32)
    for (int i = tid; i < 64 * 64; i += 256) {
        int r = i >> 6, cix = i & 63;
        int b = bBase + r, hg = hgBase + cix;
        float iv = gsm[(0 * 64 + r) * 64 + cix] + b_ih[       hg] + b_hh[       hg];
        float fv = gsm[(1 * 64 + r) * 64 + cix] + b_ih[ 512 + hg] + b_hh[ 512 + hg];
        float gv = gsm[(2 * 64 + r) * 64 + cix] + b_ih[1024 + hg] + b_hh[1024 + hg];
        float ov = gsm[(3 * 64 + r) * 64 + cix] + b_ih[1536 + hg] + b_hh[1536 + hg];
        iv = 1.0f / (1.0f + __expf(-iv));
        fv = 1.0f / (1.0f + __expf(-fv));
        gv = tanhf(gv);
        ov = 1.0f / (1.0f + __expf(-ov));
        size_t idx = (size_t)b * HG + hg;
        float c_new = fv * cbuf[idx] + iv * gv;
        float h_new = ov * tanhf(c_new);
        cbuf[idx]    = c_new;
        hout_f[idx]  = h_new;
        hout_bf[idx] = f2bf(h_new);
    }
}

// ------------------------------------------------------- attention (fp32)
__global__ __launch_bounds__(256) void attention_kernel(
    const float* __restrict__ outh,    // (Ts,B,HG)
    const float* __restrict__ att_w,   // (HG)
    const float* __restrict__ att_b,   // (1)
    float* __restrict__ att_out_f,     // (B,HG)
    unsigned short* __restrict__ att_out_bf)
{
    const int b   = blockIdx.x;
    const int tid = threadIdx.x;
    __shared__ float sc[192];
    __shared__ float red[256];

    for (int t = tid; t < Ts; t += 256) {
        float s = att_b[0];
        const float* p = outh + ((size_t)t * Bb + b) * HG;
        for (int hg = 0; hg < HG; ++hg) s += p[hg] * att_w[hg];
        sc[t] = s;
    }
    __syncthreads();

    float m = -INFINITY;
    for (int t = tid; t < Ts; t += 256) m = fmaxf(m, sc[t]);
    red[tid] = m;
    __syncthreads();
    for (int s2 = 128; s2 > 0; s2 >>= 1) {
        if (tid < s2) red[tid] = fmaxf(red[tid], red[tid + s2]);
        __syncthreads();
    }
    m = red[0];
    __syncthreads();

    float sum = 0.0f;
    for (int t = tid; t < Ts; t += 256) {
        float e = __expf(sc[t] - m);
        sc[t] = e;
        sum += e;
    }
    red[tid] = sum;
    __syncthreads();
    for (int s2 = 128; s2 > 0; s2 >>= 1) {
        if (tid < s2) red[tid] += red[tid + s2];
        __syncthreads();
    }
    const float inv = 1.0f / red[0];
    __syncthreads();

    for (int hg = tid; hg < HG; hg += 256) {
        float a = outh[((size_t)(Ts - 1) * Bb + b) * HG + hg];   // + out[:,-1,:]
        for (int t = 0; t < Ts; ++t)
            a += sc[t] * inv * outh[((size_t)t * Bb + b) * HG + hg];
        size_t idx = (size_t)b * HG + hg;
        att_out_f[idx]  = a;
        att_out_bf[idx] = f2bf(a);
    }
}

// ---------------------------------------------- output head + AR component
__global__ __launch_bounds__(256) void output_kernel(
    const float* __restrict__ hfin,   // (B,HG)
    const float* __restrict__ lg_w,   // (D,HG)
    const float* __restrict__ lg_b,   // (D)
    const float* __restrict__ x,      // (B,W,D) fp32
    const float* __restrict__ la_w,   // (AR)
    const float* __restrict__ la_b,   // (1)
    float* __restrict__ outp)         // (B,D)
{
    int idx = blockIdx.x * 256 + threadIdx.x;
    if (idx >= Bb * Dd) return;
    int b = idx / Dd, d = idx - b * Dd;
    float r = lg_b[d];
    const float* hp = hfin + (size_t)b * HG;
    const float* wp = lg_w + (size_t)d * HG;
    for (int k = 0; k < HG; ++k) r += hp[k] * wp[k];
    float a = la_b[0];
    for (int t = 0; t < AR; ++t)
        a += x[((size_t)b * Ww + (Ww - AR + t)) * Dd + d] * la_w[t];
    outp[idx] = r + a;
}

// --------------------------------------------------------------- launcher
extern "C" void kernel_launch(void* const* d_in, const int* in_sizes, int n_in,
                              void* d_out, int out_size, void* d_ws, size_t ws_size,
                              hipStream_t stream) {
    (void)in_sizes; (void)n_in; (void)out_size; (void)ws_size;
    const float* x      = (const float*)d_in[0];
    const float* conv_w = (const float*)d_in[1];
    const float* conv_b = (const float*)d_in[2];
    const float* w_ih   = (const float*)d_in[3];
    const float* w_hh   = (const float*)d_in[4];
    const float* b_ih   = (const float*)d_in[5];
    const float* b_hh   = (const float*)d_in[6];
    const float* att_w  = (const float*)d_in[7];
    const float* att_b  = (const float*)d_in[8];
    const float* lg_w   = (const float*)d_in[9];
    const float* lg_b   = (const float*)d_in[10];
    const float* la_w   = (const float*)d_in[11];
    const float* la_b   = (const float*)d_in[12];
    float* outp = (float*)d_out;

    char* ws = (char*)d_ws;
    size_t off = 0;
    auto alloc = [&](size_t bytes) -> char* {
        char* p = ws + off;
        off = (off + bytes + 255) & ~(size_t)255;
        return p;
    };

    const size_t nX   = (size_t)Bb * Ww * Dd;     // 27,611,136
    const size_t nCWP = (size_t)HC * KTP;         // 999,424 (padded)
    const size_t nWih = (size_t)4 * HG * HC;      // 1,048,576
    const size_t nHB  = (size_t)Bb * HG;          // 262,144
    const size_t nG   = (size_t)Ts * Bb * HC;     // 43,778,048

    unsigned short* xbf    = (unsigned short*)alloc(nX   * 2);
    unsigned short* cwbf   = (unsigned short*)alloc(nCWP * 2);
    unsigned short* wihbf  = (unsigned short*)alloc(nWih * 2);
    unsigned short* whhbf  = (unsigned short*)alloc(nWih * 2);
    unsigned short* gbf    = (unsigned short*)alloc(nG   * 2);
    float*          outh   = (float*)         alloc(nG   * 4);
    unsigned short* hb0    = (unsigned short*)alloc(nHB  * 2);
    unsigned short* hb1    = (unsigned short*)alloc(nHB  * 2);
    float*          cbuf   = (float*)         alloc(nHB  * 4);
    float*          attf   = (float*)         alloc(nHB  * 4);
    unsigned short* attbf  = (unsigned short*)alloc(nHB  * 2);
    float*          hfin   = (float*)         alloc(nHB  * 4);
    unsigned short* hb[2]  = { hb0, hb1 };

    auto blocks = [](size_t n) { return (unsigned)((n + 255) / 256); };

    // 1) precision conversion + state init
    cvt_f32_bf16<<<blocks(nX),   256, 0, stream>>>(x,    xbf,   (int)nX);
    cvt_convw   <<<blocks(nCWP), 256, 0, stream>>>(conv_w, cwbf);
    cvt_f32_bf16<<<blocks(nWih), 256, 0, stream>>>(w_ih, wihbf, (int)nWih);
    cvt_f32_bf16<<<blocks(nWih), 256, 0, stream>>>(w_hh, whhbf, (int)nWih);
    zero_bf16   <<<blocks(nHB),  256, 0, stream>>>(hb0,  (int)nHB);
    zero_f32    <<<blocks(nHB),  256, 0, stream>>>(cbuf, (int)nHB);

    // 2) conv im2col GEMM: M = 167*512, tiles 128x64
    conv_gemm_kernel<<<dim3((unsigned)(Ts * Bb / 128), HC / 64), 256, 0, stream>>>(
        xbf, cwbf, conv_b, gbf);

    // 3) sequential LSTM scan: one WMMA GEMM kernel per timestep
    for (int t = 0; t < Ts; ++t) {
        lstm_step_kernel<<<dim3(Bb / 64, HG / 64), 256, 0, stream>>>(
            gbf + (size_t)t * Bb * HC,
            hb[t & 1],
            wihbf, whhbf, b_ih, b_hh,
            cbuf,
            outh + (size_t)t * Bb * HG,
            hb[(t + 1) & 1]);
    }

    // 4) attention over hidden states
    attention_kernel<<<Bb, 256, 0, stream>>>(outh, att_w, att_b, attf, attbf);

    // 5) final LSTM cell on attention output (carry h=hb[Ts&1], c=cbuf)
    lstm_step_kernel<<<dim3(Bb / 64, HG / 64), 256, 0, stream>>>(
        attbf, hb[Ts & 1], wihbf, whhbf, b_ih, b_hh, cbuf, hfin, hb[(Ts + 1) & 1]);

    // 6) linear head + autoregressive component
    output_kernel<<<blocks((size_t)Bb * Dd), 256, 0, stream>>>(
        hfin, lg_w, lg_b, x, la_w, la_b, outp);
}
